// SparseBlock_Conv2d_BN_ReLU_45286135169538
// MI455X (gfx1250) — matrix operand order repaired and checked by reading, not compiled, not run on hardware
//
#include <hip/hip_runtime.h>

#define NBATCH  8
#define H_IN    400
#define W_IN    400
#define C_IN    64
#define C_OUT   64
#define BSIZE   10
#define BSTRIDE 8
#define HO      398
#define WO      398
#define KTOT    576   // 9 taps * 64 cin

typedef __attribute__((ext_vector_type(16))) __bf16 bf16x16;
typedef __attribute__((ext_vector_type(8)))  __bf16 bf16x8;
typedef __attribute__((ext_vector_type(8)))  float  f32x8;

__device__ __forceinline__ unsigned short f2bf(float f) {
  unsigned u = __float_as_uint(f);
  u += 0x7FFFu + ((u >> 16) & 1u);          // round-to-nearest-even
  return (unsigned short)(u >> 16);
}

// ---- prep: weights HWIO fp32 -> bf16 [cout][tap*64+cin]; fold BN ----------
__global__ void prep_kernel(const float* __restrict__ w,
                            const float* __restrict__ convb,
                            const float* __restrict__ gamma,
                            const float* __restrict__ beta,
                            const float* __restrict__ mean,
                            const float* __restrict__ var,
                            unsigned short* __restrict__ wsW,
                            float* __restrict__ wsScale,
                            float* __restrict__ wsBias) {
  int tid = threadIdx.x;
  for (int i = tid; i < C_OUT * KTOT; i += blockDim.x) {
    int n = i / KTOT;
    int r = i - n * KTOT;                    // r = tap*64 + cin
    wsW[i] = f2bf(w[r * C_OUT + n]);         // conv_w flat: (tap*64+cin)*64 + cout
  }
  if (tid < C_OUT) {
    float s = gamma[tid] * rsqrtf(var[tid] + 1e-3f);
    wsScale[tid] = s;
    wsBias[tid]  = (convb[tid] - mean[tid]) * s + beta[tid];
  }
}

// ---- zero the scatter target ----------------------------------------------
__global__ void zero_kernel(float4* __restrict__ out, long n4) {
  long i = (long)blockIdx.x * blockDim.x + threadIdx.x;
  long stride = (long)gridDim.x * blockDim.x;
  float4 z = {0.f, 0.f, 0.f, 0.f};
  for (; i < n4; i += stride) out[i] = z;
}

// ---- main: persistent block-conv, implicit GEMM via WMMA bf16 -------------
__global__ void __launch_bounds__(128)
sbnet_conv_kernel(const float* __restrict__ x,
                  const int* __restrict__ idx, int nb,
                  const unsigned short* __restrict__ wsW,
                  const float* __restrict__ wsScale,
                  const float* __restrict__ wsBias,
                  float* __restrict__ out) {
  extern __shared__ unsigned short smem[];
  unsigned short* sW    = smem;                  // 64*576 bf16 = 73728 B
  unsigned short* sTile = smem + C_OUT * KTOT;   // 10*10*64 bf16 = 12800 B

  const int tid  = threadIdx.x;
  const int lane = tid & 31;
  const int wave = tid >> 5;                     // 0..3 : M strip
  const int nlo  = lane & 15;
  const int aK   = (lane >> 4) * 8;              // A: hi half-wave holds K+8 / K+24
  const int bK   = (lane >> 4) * 8;              // B: hi half-wave holds K+8 / K+24
  const int mOff = (lane >> 4) * 8;              // C/D: hi half-wave holds M=8..15

  // ---- stage weights into LDS once per workgroup (L2-resident source) ----
  {
    const uint4* src = (const uint4*)wsW;
    uint4* dst = (uint4*)sW;
    for (int i = tid; i < (C_OUT * KTOT) / 8; i += blockDim.x) dst[i] = src[i];
  }

  // per-lane invariant addressing
  const int m0   = wave * 16;
  const int rowA = m0 + nlo;                     // A rows: lanes 0-15 == 16-31
  const int aBase = ((rowA >> 3) * BSIZE + (rowA & 7)) * C_IN + aK;
  int bBase[4];
#pragma unroll
  for (int nt = 0; nt < 4; ++nt) bBase[nt] = (nt * 16 + nlo) * KTOT + bK;

  float scl[4], bia[4];
#pragma unroll
  for (int nt = 0; nt < 4; ++nt) {
    scl[nt] = wsScale[nt * 16 + nlo];
    bia[nt] = wsBias[nt * 16 + nlo];
  }

  for (int b = blockIdx.x; b < nb; b += gridDim.x) {
    const int bn = idx[3 * b + 0];
    const int by = idx[3 * b + 1];
    const int bx = idx[3 * b + 2];
    const float* src =
        x + (((long)bn * H_IN + by * BSTRIDE) * W_IN + bx * BSTRIDE) * C_IN;

    __syncthreads();  // prior iteration's sTile readers done (also fences sW fill)
    // gather 10x10x64 fp32 tile -> bf16 in LDS
    for (int i = tid; i < (BSIZE * BSIZE * C_IN) / 4; i += blockDim.x) {
      int e   = i * 4;
      int row = e / (BSIZE * C_IN);
      int rem = e - row * (BSIZE * C_IN);
      float4 v = *(const float4*)(src + (long)row * (W_IN * C_IN) + rem);
      uint2 p;
      p.x = (unsigned)f2bf(v.x) | ((unsigned)f2bf(v.y) << 16);
      p.y = (unsigned)f2bf(v.z) | ((unsigned)f2bf(v.w) << 16);
      *(uint2*)(&sTile[e]) = p;
    }
    __syncthreads();

    f32x8 acc[4] = {};
    for (int t = 0; t < 9; ++t) {
      const int ty = t / 3, tx = t - 3 * ty;
      const int tOff = (ty * BSIZE + tx) * C_IN;
#pragma unroll
      for (int cb = 0; cb < 64; cb += 32) {
        const unsigned short* ap = &sTile[aBase + tOff + cb];
        bf16x8 alo = *(const bf16x8*)(ap);         // K   .. K+7  (this half-wave)
        bf16x8 ahi = *(const bf16x8*)(ap + 16);    // K+16.. K+23 (this half-wave)
        bf16x16 a = __builtin_shufflevector(alo, ahi,
            0,1,2,3,4,5,6,7,8,9,10,11,12,13,14,15);
#pragma unroll
        for (int nt = 0; nt < 4; ++nt) {
          const unsigned short* bp = &sW[bBase[nt] + t * C_IN + cb];
          bf16x8 blo = *(const bf16x8*)(bp);       // K .. K+7, K+16 .. K+23 (this half-wave)
          bf16x8 bhi = *(const bf16x8*)(bp + 16);
          bf16x16 bm = __builtin_shufflevector(blo, bhi,
              0,1,2,3,4,5,6,7,8,9,10,11,12,13,14,15);
          acc[nt] = __builtin_amdgcn_wmma_f32_16x16x32_bf16(
              false, a, false, bm, (short)0, acc[nt], false, false);
        }
      }
    }

    // fused BN + ReLU epilogue, scatter 8x8x64 block
    float* obase =
        out + (((long)bn * HO + by * BSTRIDE) * WO + bx * BSTRIDE) * C_OUT;
#pragma unroll
    for (int j = 0; j < 8; ++j) {
      int m  = m0 + mOff + j;
      int py = m >> 3, px = m & 7;
      float* orow = obase + ((long)py * WO + px) * C_OUT + nlo;
#pragma unroll
      for (int nt = 0; nt < 4; ++nt) {
        float v = fmaxf(acc[nt][j] * scl[nt] + bia[nt], 0.f);
        orow[nt * 16] = v;
      }
    }
  }
}

extern "C" void kernel_launch(void* const* d_in, const int* in_sizes, int n_in,
                              void* d_out, int out_size, void* d_ws, size_t ws_size,
                              hipStream_t stream) {
  const float* x     = (const float*)d_in[0];
  const float* w     = (const float*)d_in[1];
  const float* convb = (const float*)d_in[2];
  const float* gamma = (const float*)d_in[3];
  const float* beta  = (const float*)d_in[4];
  const float* mean  = (const float*)d_in[5];
  const float* var   = (const float*)d_in[6];
  const int*   idx   = (const int*)d_in[7];
  const int    nb    = in_sizes[7] / 3;
  float* out = (float*)d_out;

  unsigned short* wsW = (unsigned short*)d_ws;
  float* wsScale = (float*)((char*)d_ws + (size_t)C_OUT * KTOT * 2);
  float* wsBias  = wsScale + C_OUT;

  prep_kernel<<<1, 256, 0, stream>>>(w, convb, gamma, beta, mean, var,
                                     wsW, wsScale, wsBias);

  long n4 = (long)out_size / 4;   // out_size = 8*398*398*64, divisible by 4
  zero_kernel<<<4096, 256, 0, stream>>>((float4*)d_out, n4);

  int grid = nb < 800 ? nb : 800;
  size_t smem = (size_t)(C_OUT * KTOT + BSIZE * BSIZE * C_IN) * sizeof(unsigned short);
  sbnet_conv_kernel<<<grid, 128, smem, stream>>>(x, idx, nb, wsW, wsScale,
                                                 wsBias, out);
}
